// ActorCritic_86397562126736
// MI455X (gfx1250) — matrix operand order repaired and checked by reading, not compiled, not run on hardware
//
#include <hip/hip_runtime.h>
#include <stdint.h>

// ---------------------------------------------------------------------------
// Types / helpers
// ---------------------------------------------------------------------------
typedef __attribute__((ext_vector_type(16))) __bf16 bf16x16;
typedef __attribute__((ext_vector_type(8)))  float  floatx8;

__device__ __forceinline__ unsigned short f2bf(float f) {
  unsigned int u = __builtin_bit_cast(unsigned int, f);
  u += 0x7FFFu + ((u >> 16) & 1u);   // round-to-nearest-even
  return (unsigned short)(u >> 16);
}
__device__ __forceinline__ float bf2f(unsigned short h) {
  unsigned int u = ((unsigned int)h) << 16;
  return __builtin_bit_cast(float, u);
}

union FragBF { bf16x16 v; unsigned int d[8]; };

__device__ __forceinline__ float waveReduce(float v) {
#pragma unroll
  for (int off = 16; off > 0; off >>= 1) v += __shfl_xor(v, off, 32);
  return v;
}

// LDS tile row stride (halfwords): 112 bytes = 16*7 -> every row 16B-aligned
// (b128-friendly) and 28*delta % 64 != 0 for delta in 1..15 (bank spread).
#define LSTR 56

// ---------------------------------------------------------------------------
// Tile stagers. Block tile: A 128x32 (As[m][k]), B 32x128 stored Bs[n][k].
// All GEMMs have K % 32 == 0 (inputs are pre-padded), so no guards anywhere.
// ---------------------------------------------------------------------------
template <typename AT>
__device__ __forceinline__ void stageA(unsigned short (*__restrict__ As)[LSTR],
                                       const AT* __restrict__ A, int lda,
                                       int mblk, int kk, int K, int tid)
{
  if constexpr (sizeof(AT) == 4) {   // fp32 -> bf16, float4 loads, b64 stores
#pragma unroll
    for (int i = 0; i < 4; ++i) {
      int idx = tid + i * 256;                 // 0..1023 quads
      int r = idx >> 3, c4 = (idx & 7) * 4;
      const float* p = (const float*)A + (size_t)(mblk + r) * lda + kk + c4;
      if (i == 0 && kk + 64 < K) __builtin_prefetch(p + 64, 0, 0);
      float4 f = *(const float4*)p;
      uint2 pk;
      pk.x = (unsigned)f2bf(f.x) | ((unsigned)f2bf(f.y) << 16);
      pk.y = (unsigned)f2bf(f.z) | ((unsigned)f2bf(f.w) << 16);
      *(uint2*)&As[r][c4] = pk;
    }
  } else {                           // bf16 source: uint4 loads, b128 stores
#pragma unroll
    for (int i = 0; i < 2; ++i) {
      int idx = tid + i * 256;                 // 0..511 octets
      int r = idx >> 2, c8 = (idx & 3) * 8;
      const unsigned short* p = (const unsigned short*)A +
                                (size_t)(mblk + r) * lda + kk + c8;
      if (i == 0 && kk + 64 < K) __builtin_prefetch(p + 64, 0, 0);
      *(uint4*)&As[r][c8] = *(const uint4*)p;
    }
  }
}

template <typename BT>
__device__ __forceinline__ void stageB(unsigned short (*__restrict__ Bs)[LSTR],
                                       const BT* __restrict__ B, int ldb,
                                       int nblk, int kk, int K, int tid)
{
  if constexpr (sizeof(BT) == 4) {   // fp32 weights [K,N]: float4 along n
#pragma unroll
    for (int i = 0; i < 4; ++i) {
      int idx = tid + i * 256;                 // 0..1023 quads
      int k = idx >> 5, n4 = (idx & 31) * 4;
      const float* p = (const float*)B + (size_t)(kk + k) * ldb + nblk + n4;
      if (i == 0 && kk + 64 < K) __builtin_prefetch(p + (size_t)64 * ldb, 0, 0);
      float4 f = *(const float4*)p;
      Bs[n4 + 0][k] = f2bf(f.x);
      Bs[n4 + 1][k] = f2bf(f.y);
      Bs[n4 + 2][k] = f2bf(f.z);
      Bs[n4 + 3][k] = f2bf(f.w);
    }
  } else {                           // bf16 [K,N]: uint4 (8 halves) along n
#pragma unroll
    for (int i = 0; i < 2; ++i) {
      int idx = tid + i * 256;                 // 0..511 octets
      int k = idx >> 4, n8 = (idx & 15) * 8;
      const unsigned short* p = (const unsigned short*)B +
                                (size_t)(kk + k) * ldb + nblk + n8;
      if (i == 0 && kk + 64 < K) __builtin_prefetch(p + (size_t)64 * ldb, 0, 0);
      uint4 q = *(const uint4*)p;
      Bs[n8 + 0][k] = (unsigned short)(q.x);
      Bs[n8 + 1][k] = (unsigned short)(q.x >> 16);
      Bs[n8 + 2][k] = (unsigned short)(q.y);
      Bs[n8 + 3][k] = (unsigned short)(q.y >> 16);
      Bs[n8 + 4][k] = (unsigned short)(q.z);
      Bs[n8 + 5][k] = (unsigned short)(q.z >> 16);
      Bs[n8 + 6][k] = (unsigned short)(q.w);
      Bs[n8 + 7][k] = (unsigned short)(q.w >> 16);
    }
  }
}

// ---------------------------------------------------------------------------
// WMMA GEMM: C[M,N] = act(A[M,K] * B[K,N] + bias[N]); bf16 MACs, f32 acc.
// Block tile 128(M) x 128(N), 8 waves, K stepped by 32, double-buffered LDS.
// Wave w: n-tiles = 32*(w&3) + {0,16}; m-tiles = 64*(w>>2) + 16*t (t<4).
// 8 WMMA per wave per K-step (each A fragment feeds two B fragments).
// Requires M%128 == 0, N%128 == 0, K%32 == 0.
// ---------------------------------------------------------------------------
template <typename AT, typename BT, typename CT, bool RELU, bool BIAS>
__global__ __launch_bounds__(256) void wmma_gemm(
    const AT* __restrict__ A, int lda,
    const BT* __restrict__ B, int ldb,
    CT* __restrict__ C, int ldc,
    const float* __restrict__ bias,
    int M, int N, int K)
{
  __shared__ unsigned short As[2][128][LSTR];
  __shared__ unsigned short Bs[2][128][LSTR];

  const int tid  = threadIdx.x;
  const int lane = tid & 31;
  const int wave = tid >> 5;
  const int wn   = wave & 3;
  const int wm   = wave >> 2;
  const int lrow = lane & 15;
  const int lh   = lane >> 4;

  const int mblk = blockIdx.y * 128;
  const int nblk = blockIdx.x * 128;

  floatx8 acc[4][2] = {{floatx8{}, floatx8{}}, {floatx8{}, floatx8{}},
                       {floatx8{}, floatx8{}}, {floatx8{}, floatx8{}}};

  const int nsteps = K >> 5;

  // prologue: stage first tile
  stageA(As[0], A, lda, mblk, 0, K, tid);
  stageB(Bs[0], B, ldb, nblk, 0, K, tid);
  __syncthreads();

  for (int s = 0; s < nsteps; ++s) {
    const int cur = s & 1;
    // stage next tile into the other buffer (overlaps with WMMA below)
    if (s + 1 < nsteps) {
      int kn = (s + 1) * 32;
      stageA(As[cur ^ 1], A, lda, mblk, kn, K, tid);
      stageB(Bs[cur ^ 1], B, ldb, nblk, kn, K, tid);
    }

    // Two B fragments: lane holds col n; VGPR v: K = lh*16 + 2v
    FragBF bf0, bf1;
#pragma unroll
    for (int v = 0; v < 8; ++v) {
      int k = lh * 16 + v * 2;
      bf0.d[v] = *(const unsigned int*)&Bs[cur][wn * 32 + lrow][k];
      bf1.d[v] = *(const unsigned int*)&Bs[cur][wn * 32 + 16 + lrow][k];
    }
    // 4 stacked M tiles; A frag: VGPR v: K = lh*16 + 2v
#pragma unroll
    for (int t = 0; t < 4; ++t) {
      FragBF af;
      const unsigned short* arow = &As[cur][wm * 64 + t * 16 + lrow][0];
#pragma unroll
      for (int v = 0; v < 8; ++v)
        af.d[v] = *(const unsigned int*)&arow[lh * 16 + v * 2];
      acc[t][0] = __builtin_amdgcn_wmma_f32_16x16x32_bf16(
          false, af.v, false, bf0.v, (short)0, acc[t][0], false, false);
      acc[t][1] = __builtin_amdgcn_wmma_f32_16x16x32_bf16(
          false, af.v, false, bf1.v, (short)0, acc[t][1], false, false);
    }
    __syncthreads();   // reads of 'cur' done; next iter may overwrite it
  }

  // epilogue: C/D layout VGPR r -> (M = r + 8*lh, N = lrow)
#pragma unroll
  for (int p = 0; p < 2; ++p) {
    const int nglob = nblk + wn * 32 + p * 16 + lrow;
    float bv = 0.f;
    if constexpr (BIAS) bv = bias[nglob];
#pragma unroll
    for (int t = 0; t < 4; ++t) {
#pragma unroll
      for (int r = 0; r < 8; ++r) {
        int mg = mblk + wm * 64 + t * 16 + lh * 8 + r;
        float v = acc[t][p][r] + bv;
        if constexpr (RELU) v = fmaxf(v, 0.f);
        if constexpr (sizeof(CT) == 2)
          ((unsigned short*)C)[(size_t)mg * ldc + nglob] = f2bf(v);
        else
          ((float*)C)[(size_t)mg * ldc + nglob] = v;
      }
    }
  }
}

// ---------------------------------------------------------------------------
// Prep: pad/convert all_embeds [4096,300] f32 -> [4096,320] bf16 (zero pad)
// One 320-thread block per row.
// ---------------------------------------------------------------------------
__global__ __launch_bounds__(320) void k_pad_embeds(
    const float* __restrict__ in, unsigned short* __restrict__ out)
{
  int col = threadIdx.x, row = blockIdx.x;
  out[(size_t)row * 320 + col] =
      (col < 300) ? f2bf(in[(size_t)row * 300 + col]) : (unsigned short)0;
}

// Prep: transpose+pad semantic_W [512,300] f32 -> Bt [320,512] bf16
__global__ __launch_bounds__(256) void k_prep_semW(
    const float* __restrict__ W, unsigned short* __restrict__ Bt)
{
  int idx = blockIdx.x * 256 + threadIdx.x;   // 0 .. 320*512-1
  int k = idx >> 9, n = idx & 511;
  Bt[idx] = (k < 300) ? f2bf(W[(size_t)n * 300 + k]) : (unsigned short)0;
}

// ---------------------------------------------------------------------------
// Batch-1 heads: visual (K=8192), semantic (K=300), scores512 (K=1000).
// One wave per output neuron.
// ---------------------------------------------------------------------------
__global__ __launch_bounds__(256) void k_heads1(
    const float* __restrict__ frames, const float* __restrict__ scores,
    const float* __restrict__ word_embed,
    const float* __restrict__ vW, const float* __restrict__ vb,
    const float* __restrict__ sW, const float* __restrict__ sb,
    const float* __restrict__ scW, const float* __restrict__ scb,
    float* __restrict__ visual, float* __restrict__ semantic,
    float* __restrict__ scores512)
{
  int wave = threadIdx.x >> 5, lane = threadIdx.x & 31;
  int gw = blockIdx.x * 8 + wave;             // 0..1535
  const float* w; const float* x; int K; float b; float* out; int o;
  if (gw < 512)        { o = gw;        w = vW  + (size_t)o * 8192; x = frames;     K = 8192; b = vb[o];  out = visual;    }
  else if (gw < 1024)  { o = gw - 512;  w = sW  + (size_t)o * 300;  x = word_embed; K = 300;  b = sb[o];  out = semantic;  }
  else                 { o = gw - 1024; w = scW + (size_t)o * 1000; x = scores;     K = 1000; b = scb[o]; out = scores512; }
  float acc = 0.f;
  for (int k = lane; k < K; k += 32) acc += w[k] * x[k];
  acc = waveReduce(acc);
  if (lane == 0) out[o] = fmaxf(acc + b, 0.f);
}

// nodes[:, 0:512] = bf16(scores512) broadcast over 4096 rows
__global__ __launch_bounds__(256) void k_fill_left(
    const float* __restrict__ scores512, unsigned short* __restrict__ nodes)
{
  int idx = blockIdx.x * 256 + threadIdx.x;   // 0 .. 4096*512-1
  int row = idx >> 9, col = idx & 511;
  nodes[(size_t)row * 1024 + col] = f2bf(scores512[col]);
}

// t3[m] = dot(h2_bf16[m, 0:1024], gc3_W[:,0])  (one wave per row)
__global__ __launch_bounds__(256) void k_gemv_col(
    const unsigned short* __restrict__ h, const float* __restrict__ w,
    float* __restrict__ t3)
{
  int wave = threadIdx.x >> 5, lane = threadIdx.x & 31;
  int row = blockIdx.x * 8 + wave;
  const unsigned short* hr = h + (size_t)row * 1024;
  float acc = 0.f;
  for (int k = lane; k < 1024; k += 32) acc += bf2f(hr[k]) * w[k];
  acc = waveReduce(acc);
  if (lane == 0) t3[row] = acc;
}

// h3[m] = relu(dot(adj[m,:], t3) + gc3_b)  (one 256-thread block per row)
__global__ __launch_bounds__(256) void k_adj_vec(
    const float* __restrict__ adj, const float* __restrict__ t3,
    const float* __restrict__ b, float* __restrict__ h3)
{
  __shared__ float red[256];
  int tid = threadIdx.x, row = blockIdx.x;
  const float4* ar = (const float4*)(adj + (size_t)row * 4096);
  const float4* tv = (const float4*)t3;
  float acc = 0.f;
  for (int q = tid; q < 1024; q += 256) {
    float4 a = ar[q], t = tv[q];
    acc += a.x * t.x + a.y * t.y + a.z * t.z + a.w * t.w;
  }
  red[tid] = acc; __syncthreads();
#pragma unroll
  for (int s = 128; s > 0; s >>= 1) {
    if (tid < s) red[tid] += red[tid + s];
    __syncthreads();
  }
  if (tid == 0) h3[row] = fmaxf(red[0] + b[0], 0.f);
}

// gcn[o] = relu(dot(gcn512_W[o,:4096], h3) + b[o])
__global__ __launch_bounds__(256) void k_gcn512(
    const float* __restrict__ W, const float* __restrict__ b,
    const float* __restrict__ h3, float* __restrict__ gcn)
{
  int wave = threadIdx.x >> 5, lane = threadIdx.x & 31;
  int o = blockIdx.x * 8 + wave;
  const float* w = W + (size_t)o * 4096;
  float acc = 0.f;
  for (int k = lane; k < 4096; k += 32) acc += w[k] * h3[k];
  acc = waveReduce(acc);
  if (lane == 0) gcn[o] = fmaxf(acc + b[o], 0.f);
}

// x[o] = relu(dot(hidden_W[o,:1536], joint) + b[o]); joint = [visual|semantic|gcn]
__global__ __launch_bounds__(256) void k_hidden(
    const float* __restrict__ visual, const float* __restrict__ semantic,
    const float* __restrict__ gcn, const float* __restrict__ W,
    const float* __restrict__ b, float* __restrict__ x)
{
  int wave = threadIdx.x >> 5, lane = threadIdx.x & 31;
  int o = blockIdx.x * 8 + wave;
  const float* w = W + (size_t)o * 1536;
  float acc = 0.f;
  for (int k = lane; k < 1536; k += 32) {
    float j = (k < 512) ? visual[k] : (k < 1024) ? semantic[k - 512] : gcn[k - 1024];
    acc += w[k] * j;
  }
  acc = waveReduce(acc);
  if (lane == 0) x[o] = fmaxf(acc + b[o], 0.f);
}

// out[0] = critic; out[1..6] = actor logits (no activation)
__global__ __launch_bounds__(256) void k_final(
    const float* __restrict__ x, const float* __restrict__ cW,
    const float* __restrict__ cb, const float* __restrict__ aW,
    const float* __restrict__ ab, float* __restrict__ out)
{
  int wave = threadIdx.x >> 5, lane = threadIdx.x & 31;
  if (wave >= 7) return;
  const float* w; float b; int oi;
  if (wave == 0) { w = cW; b = cb[0]; oi = 0; }
  else           { w = aW + (size_t)(wave - 1) * 512; b = ab[wave - 1]; oi = wave; }
  float acc = 0.f;
  for (int k = lane; k < 512; k += 32) acc += w[k] * x[k];
  acc = waveReduce(acc);
  if (lane == 0) out[oi] = acc + b;
}

// ---------------------------------------------------------------------------
// Launch
// ---------------------------------------------------------------------------
extern "C" void kernel_launch(void* const* d_in, const int* in_sizes, int n_in,
                              void* d_out, int out_size, void* d_ws, size_t ws_size,
                              hipStream_t stream)
{
  const float* frames     = (const float*)d_in[0];
  const float* scores     = (const float*)d_in[1];
  const float* word_embed = (const float*)d_in[2];
  const float* all_embeds = (const float*)d_in[3];
  const float* adj        = (const float*)d_in[4];
  const float* visual_W   = (const float*)d_in[5];
  const float* visual_b   = (const float*)d_in[6];
  const float* semantic_W = (const float*)d_in[7];
  const float* semantic_b = (const float*)d_in[8];
  const float* score_W    = (const float*)d_in[9];
  const float* score_b    = (const float*)d_in[10];
  const float* gc1_W      = (const float*)d_in[11];
  const float* gc1_b      = (const float*)d_in[12];
  const float* gc2_W      = (const float*)d_in[13];
  const float* gc2_b      = (const float*)d_in[14];
  const float* gc3_W      = (const float*)d_in[15];
  const float* gc3_b      = (const float*)d_in[16];
  const float* gcn512_W   = (const float*)d_in[17];
  const float* gcn512_b   = (const float*)d_in[18];
  const float* hidden_W   = (const float*)d_in[19];
  const float* hidden_b   = (const float*)d_in[20];
  const float* critic_W   = (const float*)d_in[21];
  const float* critic_b   = (const float*)d_in[22];
  const float* actor_W    = (const float*)d_in[23];
  const float* actor_b    = (const float*)d_in[24];

  // workspace layout (~28.2 MB)
  float* F          = (float*)d_ws;
  float* visual     = F;            // 512
  float* semantic   = F + 512;      // 512
  float* scores512  = F + 1024;     // 512
  float* t3         = F + 1536;     // 4096
  float* h3         = F + 5632;     // 4096
  float* gcn        = F + 9728;     // 512
  float* xbuf       = F + 10240;    // 512
  unsigned short* U     = (unsigned short*)(F + 10752);   // 16B-aligned
  unsigned short* nodes = U;                              // 4096x1024 bf16
  unsigned short* t1    = nodes + (size_t)4096 * 1024;    // 4096x1024 bf16
  unsigned short* h1    = t1    + (size_t)4096 * 1024;    // 4096x1024 bf16
  unsigned short* embp  = h1    + (size_t)4096 * 1024;    // 4096x320 bf16
  unsigned short* semWt = embp  + (size_t)4096 * 320;     // 320x512 bf16
  unsigned short* t2    = nodes;                          // reuse (nodes dead)
  unsigned short* h2    = t1;                             // reuse (t1 dead)

  // batch-1 heads (independent of GCN path)
  k_heads1<<<192, 256, 0, stream>>>(frames, scores, word_embed,
                                    visual_W, visual_b, semantic_W, semantic_b,
                                    score_W, score_b, visual, semantic, scores512);
  // pad/convert GEMM inputs once (removes all tail guards from hot loops)
  k_pad_embeds<<<4096, 320, 0, stream>>>(all_embeds, embp);
  k_prep_semW<<<640, 256, 0, stream>>>(semantic_W, semWt);
  // nodes[:,0:512] = scores512 broadcast
  k_fill_left<<<8192, 256, 0, stream>>>(scores512, nodes);
  // nodes[:,512:1024] = relu(embp @ semWt + semantic_b)   (K=320)
  wmma_gemm<unsigned short, unsigned short, unsigned short, true, true>
      <<<dim3(4, 32), 256, 0, stream>>>(embp, 320, semWt, 512,
                                        nodes + 512, 1024, semantic_b,
                                        4096, 512, 320);
  // gc1: t1 = nodes @ gc1_W ; h1 = relu(adj @ t1 + gc1_b)
  wmma_gemm<unsigned short, float, unsigned short, false, false>
      <<<dim3(8, 32), 256, 0, stream>>>(nodes, 1024, gc1_W, 1024, t1, 1024,
                                        nullptr, 4096, 1024, 1024);
  wmma_gemm<float, unsigned short, unsigned short, true, true>
      <<<dim3(8, 32), 256, 0, stream>>>(adj, 4096, t1, 1024, h1, 1024,
                                        gc1_b, 4096, 1024, 4096);
  // gc2: t2 = h1 @ gc2_W ; h2 = relu(adj @ t2 + gc2_b)
  wmma_gemm<unsigned short, float, unsigned short, false, false>
      <<<dim3(8, 32), 256, 0, stream>>>(h1, 1024, gc2_W, 1024, t2, 1024,
                                        nullptr, 4096, 1024, 1024);
  wmma_gemm<float, unsigned short, unsigned short, true, true>
      <<<dim3(8, 32), 256, 0, stream>>>(adj, 4096, t2, 1024, h2, 1024,
                                        gc2_b, 4096, 1024, 4096);
  // gc3 (N=1): t3 = h2 @ gc3_W ; h3 = relu(adj @ t3 + gc3_b)
  k_gemv_col<<<512, 256, 0, stream>>>(h2, gc3_W, t3);
  k_adj_vec<<<4096, 256, 0, stream>>>(adj, t3, gc3_b, h3);
  // gcn_512, hidden, critic/actor
  k_gcn512<<<64, 256, 0, stream>>>(gcn512_W, gcn512_b, h3, gcn);
  k_hidden<<<64, 256, 0, stream>>>(visual, semantic, gcn, hidden_W, hidden_b, xbuf);
  k_final<<<1, 256, 0, stream>>>(xbuf, critic_W, critic_b, actor_W, actor_b,
                                 (float*)d_out);
}
